// TDSSeq2Seq_79499844649549
// MI455X (gfx1250) — compile-verified, matrix operand order checked by reading, not run on hardware
//
#include <hip/hip_runtime.h>
#include <hip/hip_bf16.h>
#include <stdint.h>

// ---------------------------------------------------------------------------
// Persistent cooperative attention-decoder for MI455X (gfx1250, wave32).
//  - encoding (keys||values) converted once to bf16 in workspace (64 MB,
//    L2-resident alongside fp32 weights -> every step streams from L2)
//  - scores via v_wmma_f32_16x16x32_bf16 (A = 16 key rows, B = h broadcast)
//  - block-local online softmax + partial attn@V, merged by block 0
//  - grid-wide sync via monotonic atomic barrier (zeroed by init kernel)
// ---------------------------------------------------------------------------

#define T_ENC    32768
#define KEY_DIM  512
#define VAL_DIM  512
#define HID      512
#define EMB      512
#define VOCAB    31
#define EOS_ID   29
#define MAX_LEN  200
#define NBLK     128
#define NTHR     256
#define RPB      (T_ENC / NBLK)      // 256 rows (keys) per block
#define LOGIT_N  (MAX_LEN * VOCAB)   // 6200
#define ATTN_OFF (LOGIT_N + 1)       // lens (1 int) sits between
#define ATT_SCALE 0.04419417382415922f   // 1/sqrt(512)

typedef __attribute__((ext_vector_type(16))) __bf16 v16bf;
typedef __attribute__((ext_vector_type(8)))  float  v8f;
typedef __attribute__((ext_vector_type(4)))  unsigned int vu4;

union FragBF { vu4 u[2]; v16bf v; };

__device__ __forceinline__ unsigned short f32_to_bf16_rne(float f) {
    unsigned u = __float_as_uint(f);
    u += 0x7FFFu + ((u >> 16) & 1u);
    return (unsigned short)(u >> 16);
}
__device__ __forceinline__ float bf16lo(unsigned int u) { return __uint_as_float(u << 16); }
__device__ __forceinline__ float bf16hi(unsigned int u) { return __uint_as_float(u & 0xFFFF0000u); }

// Monotonic grid barrier: counter only ever increments within one launch.
__device__ __forceinline__ void grid_sync(int* ctr, int& epoch) {
    epoch++;
    __threadfence();
    __syncthreads();
    if (threadIdx.x == 0) {
        __hip_atomic_fetch_add(ctr, 1, __ATOMIC_RELEASE, __HIP_MEMORY_SCOPE_AGENT);
        while (__hip_atomic_load(ctr, __ATOMIC_ACQUIRE, __HIP_MEMORY_SCOPE_AGENT) < epoch * NBLK)
            __builtin_amdgcn_s_sleep(2);
    }
    __syncthreads();
    __threadfence();
}

__global__ void seq2seq_init_kernel(int* ctr, float* ybuf, float* hbuf,
                                    const float* __restrict__ hidden,
                                    const float* __restrict__ embed,
                                    int* out_lens) {
    const int tid = threadIdx.x;
    if (tid == 0) { *ctr = 0; out_lens[0] = MAX_LEN; }
    for (int c = tid; c < HID; c += blockDim.x) {
        ybuf[c] = embed[EOS_ID * EMB + c];   // y0 = embed_table[EOS]
        hbuf[c] = hidden[c];                 // h0
    }
}

__global__ __launch_bounds__(NTHR, 1)
void seq2seq_decoder_kernel(const float* __restrict__ enc,
                            const float* __restrict__ embed,
                            const float* __restrict__ w_ih,
                            const float* __restrict__ w_hh,
                            const float* __restrict__ b_ih,
                            const float* __restrict__ b_hh,
                            const float* __restrict__ w_out,
                            const float* __restrict__ b_out,
                            float* __restrict__ out,
                            int* ctr, float* ybuf, float* hbuf, float* msg,
                            float* part_m, float* part_s, float* vpart,
                            unsigned short* kbf, unsigned short* vbf) {
    const int tid  = threadIdx.x;
    const int b    = blockIdx.x;
    const int lane = tid & 31;
    const int wv   = tid >> 5;
    int epoch = 0;

    __shared__ float s_sc[RPB];                     // raw scores * scale
    __shared__ float s_e[RPB];                      // exp(score - m_local)
    __shared__ __align__(16) unsigned short s_hbf[HID];   // h' in bf16
    __shared__ float s_red[8];
    __shared__ float s_mloc;
    __shared__ float s_sum[HID];                    // block0: summary
    __shared__ float s_logit[VOCAB];
    __shared__ int   s_best;

    // ---- one-time conversion: fp32 encoding -> bf16 keys / values ----
    {
        const int gtid = b * NTHR + tid;
        for (int i = gtid; i < T_ENC * 1024; i += NBLK * NTHR) {
            const int t = i >> 10, j = i & 1023;
            const unsigned short v = f32_to_bf16_rne(enc[i]);
            if (j < KEY_DIM) kbf[t * KEY_DIM + j] = v;
            else             vbf[t * VAL_DIM + (j - KEY_DIM)] = v;
        }
    }
    grid_sync(ctr, epoch);

    const int t0 = b * RPB;

    for (int step = 0; step < MAX_LEN; ++step) {
        float* hcur = hbuf + ((step & 1) * HID);
        float* hnew = hbuf + (((step + 1) & 1) * HID);

        // ================= P1: flush prev attn row + GRU ==================
        if (step > 0) {
            const float fac = expf(s_mloc - msg[0]) / msg[1];
            out[ATTN_OFF + (size_t)(step - 1) * T_ENC + t0 + tid] = s_e[tid] * fac;
        }
        if (wv < 4) {   // 128 blocks * 4 waves = 512 hidden dims, 1 per wave
            const int dim = b * 4 + wv;
            float p0 = 0.f, p1 = 0.f, p2 = 0.f, p3 = 0.f, p4 = 0.f, p5 = 0.f;
            for (int k = lane; k < HID; k += 32) {
                const float xv = ybuf[k], hv = hcur[k];
                p0 = fmaf(xv, w_ih[(0 * HID + dim) * EMB + k], p0);
                p1 = fmaf(xv, w_ih[(1 * HID + dim) * EMB + k], p1);
                p2 = fmaf(xv, w_ih[(2 * HID + dim) * EMB + k], p2);
                p3 = fmaf(hv, w_hh[(0 * HID + dim) * HID + k], p3);
                p4 = fmaf(hv, w_hh[(1 * HID + dim) * HID + k], p4);
                p5 = fmaf(hv, w_hh[(2 * HID + dim) * HID + k], p5);
            }
            #pragma unroll
            for (int o = 16; o; o >>= 1) {
                p0 += __shfl_down(p0, o); p1 += __shfl_down(p1, o);
                p2 += __shfl_down(p2, o); p3 += __shfl_down(p3, o);
                p4 += __shfl_down(p4, o); p5 += __shfl_down(p5, o);
            }
            if (lane == 0) {
                const float ir = p0 + b_ih[dim],           hr = p3 + b_hh[dim];
                const float iz = p1 + b_ih[HID + dim],     hz = p4 + b_hh[HID + dim];
                const float in_ = p2 + b_ih[2 * HID + dim], hn = p5 + b_hh[2 * HID + dim];
                const float r = 1.f / (1.f + expf(-(ir + hr)));
                const float z = 1.f / (1.f + expf(-(iz + hz)));
                const float n = tanhf(in_ + r * hn);
                hnew[dim] = (1.f - z) * n + z * hcur[dim];
            }
        }
        grid_sync(ctr, epoch);

        // ============ P2: WMMA scores + online softmax + attn@V ===========
        for (int c = tid; c < HID; c += NTHR) s_hbf[c] = f32_to_bf16_rne(hnew[c]);
        __syncthreads();

        // A lane layout (16-bit, 16x32): lanes 0-15 hold K{0..7,16..23} of
        // row M=lane; lanes 16-31 hold K{8..15,24..31} of row M=lane-16.
        // B (32x16) uses the same K striping per lane; h broadcast over N.
        const int koff = (lane < 16) ? 0 : 8;
        const vu4* kb4 = (const vu4*)kbf;
        const vu4* hb4 = (const vu4*)s_hbf;
        for (int j = 0; j < 2; ++j) {
            const int tt  = wv * 2 + j;                    // 16 tiles of 16 rows
            const int row = t0 + tt * 16 + (lane & 15);
            v8f acc = {};
            #pragma unroll
            for (int kk = 0; kk < 16; ++kk) {
                const int k0 = kk * 32;
                FragBF a, bm;
                a.u[0]  = kb4[(row * KEY_DIM + k0 + koff) >> 3];
                a.u[1]  = kb4[(row * KEY_DIM + k0 + koff + 16) >> 3];
                bm.u[0] = hb4[(k0 + koff) >> 3];
                bm.u[1] = hb4[(k0 + koff + 16) >> 3];
                acc = __builtin_amdgcn_wmma_f32_16x16x32_bf16(
                        false, a.v, false, bm.v, (short)0, acc, false, false);
            }
            // D layout: VGPR r, lanes 0-15 -> M=r ; lanes 16-31 -> M=8+r.
            if (lane == 0 || lane == 16) {
                const int mb = tt * 16 + (lane >> 4) * 8;
                #pragma unroll
                for (int m = 0; m < 8; ++m) s_sc[mb + m] = acc[m] * ATT_SCALE;
            }
        }
        __syncthreads();

        // local max over this block's 256 scores
        float v = s_sc[tid];
        #pragma unroll
        for (int o = 16; o; o >>= 1) v = fmaxf(v, __shfl_down(v, o));
        if (lane == 0) s_red[wv] = v;
        __syncthreads();
        if (tid == 0) {
            float m = s_red[0];
            #pragma unroll
            for (int i = 1; i < 8; ++i) m = fmaxf(m, s_red[i]);
            s_mloc = m;
        }
        __syncthreads();
        const float ml = s_mloc;
        const float e0 = expf(s_sc[tid] - ml);
        s_e[tid] = e0;
        float sv = e0;
        #pragma unroll
        for (int o = 16; o; o >>= 1) sv += __shfl_down(sv, o);
        if (lane == 0) s_red[wv] = sv;
        __syncthreads();
        if (tid == 0) {
            float s2 = 0.f;
            #pragma unroll
            for (int i = 0; i < 8; ++i) s2 += s_red[i];
            part_m[b] = ml; part_s[b] = s2;
        }
        __syncthreads();

        // partial summary: thread owns columns 2*tid, 2*tid+1 (packed bf16)
        {
            const unsigned int* vr = (const unsigned int*)vbf + (size_t)t0 * (VAL_DIM / 2) + tid;
            float A0 = 0.f, A1 = 0.f;
            for (int r = 0; r < RPB; ++r) {
                const unsigned int u = vr[(size_t)r * (VAL_DIM / 2)];
                const float e = s_e[r];
                A0 = fmaf(e, bf16lo(u), A0);
                A1 = fmaf(e, bf16hi(u), A1);
            }
            vpart[b * VAL_DIM + 2 * tid]     = A0;
            vpart[b * VAL_DIM + 2 * tid + 1] = A1;
        }
        grid_sync(ctr, epoch);

        // ====== P3 (block 0): merge partials, logits, argmax, y_next ======
        if (b == 0) {
            if (tid == 0) {
                float m = part_m[0];
                for (int i = 1; i < NBLK; ++i) m = fmaxf(m, part_m[i]);
                float S = 0.f;
                for (int i = 0; i < NBLK; ++i) S += part_s[i] * expf(part_m[i] - m);
                msg[0] = m; msg[1] = S;
            }
            __syncthreads();
            const float mg = msg[0], Sg = msg[1];
            float a0 = 0.f, a1 = 0.f;
            for (int i = 0; i < NBLK; ++i) {
                const float sc = expf(part_m[i] - mg);
                a0 = fmaf(vpart[i * VAL_DIM + tid],       sc, a0);
                a1 = fmaf(vpart[i * VAL_DIM + tid + 256], sc, a1);
            }
            s_sum[tid]       = a0 / Sg;
            s_sum[tid + 256] = a1 / Sg;
            __syncthreads();
            // logits = [summary, h'] @ w_out + b_out  (one logit per wave slot)
            for (int l = wv; l < VOCAB; l += 8) {
                float p = 0.f;
                for (int k = lane; k < VAL_DIM + HID; k += 32) {
                    const float ck = (k < VAL_DIM) ? s_sum[k] : hnew[k - VAL_DIM];
                    p = fmaf(ck, w_out[k * VOCAB + l], p);
                }
                #pragma unroll
                for (int o = 16; o; o >>= 1) p += __shfl_down(p, o);
                if (lane == 0) {
                    const float lg = p + b_out[l];
                    s_logit[l] = lg;
                    out[step * VOCAB + l] = lg;
                }
            }
            __syncthreads();
            if (tid == 0) {
                int best = 0; float bv = s_logit[0];
                for (int l = 1; l < VOCAB; ++l)
                    if (s_logit[l] > bv) { bv = s_logit[l]; best = l; }
                s_best = best;
            }
            __syncthreads();
            const int bestt = s_best;
            for (int c = tid; c < EMB; c += NTHR) ybuf[c] = embed[bestt * EMB + c];
        }
        grid_sync(ctr, epoch);
    }

    // flush final step's attention row
    {
        const float fac = expf(s_mloc - msg[0]) / msg[1];
        out[ATTN_OFF + (size_t)(MAX_LEN - 1) * T_ENC + t0 + tid] = s_e[tid] * fac;
    }
}

extern "C" void kernel_launch(void* const* d_in, const int* in_sizes, int n_in,
                              void* d_out, int out_size, void* d_ws, size_t ws_size,
                              hipStream_t stream) {
    const float* enc   = (const float*)d_in[0];
    const float* hid   = (const float*)d_in[1];
    const float* emb   = (const float*)d_in[2];
    const float* w_ih  = (const float*)d_in[3];
    const float* w_hh  = (const float*)d_in[4];
    const float* b_ih  = (const float*)d_in[5];
    const float* b_hh  = (const float*)d_in[6];
    const float* w_out = (const float*)d_in[7];
    const float* b_out = (const float*)d_in[8];
    float* out = (float*)d_out;

    // workspace carve (needs ~64.3 MB: bf16 copy of encoding + partials)
    uintptr_t base = (uintptr_t)d_ws;
    int*   ctr    = (int*)base;
    float* ybuf   = (float*)(base + 256);
    float* hbuf   = ybuf + EMB;            // 2 * HID (ping-pong h)
    float* msg    = hbuf + 2 * HID;        // [m_global, S_global]
    float* part_m = msg + 2;               // NBLK
    float* part_s = part_m + NBLK;         // NBLK
    float* vpart  = part_s + NBLK;         // NBLK * VAL_DIM
    uintptr_t kptr = ((uintptr_t)(vpart + NBLK * VAL_DIM) + 255u) & ~(uintptr_t)255u;
    unsigned short* kbf = (unsigned short*)kptr;                 // 32 MB
    unsigned short* vbf = kbf + (size_t)T_ENC * KEY_DIM;         // 32 MB

    seq2seq_init_kernel<<<1, 512, 0, stream>>>(ctr, ybuf, hbuf, hid, emb,
                                               ((int*)d_out) + LOGIT_N);
    seq2seq_decoder_kernel<<<NBLK, NTHR, 0, stream>>>(
        enc, emb, w_ih, w_hh, b_ih, b_hh, w_out, b_out, out,
        ctr, ybuf, hbuf, msg, part_m, part_s, vpart, kbf, vbf);
}
